// SeqGGNN_59210419143210
// MI455X (gfx1250) — compile-verified, matrix operand order checked by reading, not run on hardware
//
#include <hip/hip_runtime.h>
#include <hip/hip_bf16.h>

// Problem geometry (fixed by the reference)
#define GG_H 256
#define GG_B 64
#define GG_S 512
#define GG_V 32000
#define GG_BS (GG_B * GG_S)       // 32768 rows of h
#define GG_STEPS 10
#define GG_KT (GG_H / 32)         // 8 k-tiles of 32 (K=256)
#define GG_NT_H (GG_H / 16)       // 16 column tiles over H
#define GG_MT (GG_BS / 16)        // 2048 row tiles
#define GG_FRAG_H 4096            // halves per packed n-tile (8 kt * 32 lanes * 16)

typedef __attribute__((ext_vector_type(16))) __bf16 v16bf;
typedef __attribute__((ext_vector_type(8)))  __bf16 v8bf;
typedef __attribute__((ext_vector_type(8)))  float  v8f;
typedef int v4i __attribute__((vector_size(16)));

#define GG_GLOBAL_AS __attribute__((address_space(1)))
#define GG_LDS_AS    __attribute__((address_space(3)))

// ---------------------------------------------------------------------------
// D = A(16x32 bf16) x B(32x16 bf16) + C(16x16 f32), wave32 WMMA
// ---------------------------------------------------------------------------
__device__ __forceinline__ v8f wmma_bf16(v16bf a, v16bf b, v8f c) {
#if defined(__gfx1250__)
  return __builtin_amdgcn_wmma_f32_16x16x32_bf16(false, a, false, b, (short)0, c,
                                                 false, false);
#else
  return c;  // host / non-gfx1250 semantic pass only
#endif
}

// 16-byte async global -> LDS copy (CDNA5 GLOBAL_LOAD_ASYNC_TO_LDS_B128,
// tracked by ASYNCcnt). Prototype (from compiler diagnostics): param0 is a
// non-const int4* in the global address space; param1 is the LDS-side pointer.
__device__ __forceinline__ void cp16_g2l(const __bf16* g, __bf16* l) {
#if defined(__gfx1250__) && __has_builtin(__builtin_amdgcn_global_load_async_to_lds_b128)
  __bf16* gnc = const_cast<__bf16*>(g);
  __builtin_amdgcn_global_load_async_to_lds_b128(
      (GG_GLOBAL_AS v4i*)gnc, (GG_LDS_AS v4i*)l, 0, 0);
#else
  *(v8bf*)l = *(const v8bf*)g;
#endif
}

__device__ __forceinline__ void cp_wait_all() {
#if defined(__gfx1250__) && __has_builtin(__builtin_amdgcn_global_load_async_to_lds_b128)
#if __has_builtin(__builtin_amdgcn_s_wait_asynccnt)
  __builtin_amdgcn_s_wait_asynccnt(0);
#else
  asm volatile("s_wait_asynccnt 0x0" ::: "memory");
#endif
#endif
  __syncthreads();
}

// Stage R packed n-tile regions (8 KB each) into LDS, 128 threads cooperating.
template <int R>
__device__ __forceinline__ void stage_regions(const __bf16* const (&src)[R],
                                              __bf16* sB, int tid) {
#pragma unroll
  for (int r = 0; r < R; ++r) {
#pragma unroll
    for (int i = 0; i < 4; ++i) {
      int off = (i * 128 + tid) * 8;  // halves: 128 thr x 16 B per pass
      cp16_g2l(src[r] + off, sB + r * GG_FRAG_H + off);
    }
  }
}

// A fragment (16x32 bf16) from a row-major (rows x GG_H) bf16 matrix.
// ISA layout: lane L holds row M = L&15; lanes 0-15 hold K {0..7,16..23},
// lanes 16-31 hold K {8..15,24..31}  -> two 16-byte chunks per lane.
__device__ __forceinline__ v16bf ld_a_frag(const __bf16* A, int row, int k0,
                                           int lane, bool valid) {
  v16bf f;
  if (valid) {
    int khalf = (lane < 16) ? 0 : 8;
    const __bf16* p = A + (size_t)row * GG_H + k0 + khalf;
    v8bf c0 = *(const v8bf*)(p);
    v8bf c1 = *(const v8bf*)(p + 16);
#pragma unroll
    for (int j = 0; j < 8; ++j) { f[j] = c0[j]; f[j + 8] = c1[j]; }
  } else {
#pragma unroll
    for (int j = 0; j < 16; ++j) f[j] = (__bf16)0.0f;
  }
  return f;
}

// B fragment from an LDS-staged region: one 32 B read per lane (ds_load_b128).
__device__ __forceinline__ v16bf ld_b_lds(const __bf16* sB, int r, int kt,
                                          int lane) {
  return *(const v16bf*)(sB + ((size_t)(r * GG_KT + kt) * 32 + lane) * 16);
}

// ---------------------------------------------------------------------------
// Repack fp32 weights -> bf16 B-fragment layout.
// mode 0: src = W_edge (3,H,H); logical B[k, e*H+c] = W_edge[e][k][c]
// mode 1: src = (N,H) row-major; logical B[k,n] = src[n][k]  (transpose)
// ---------------------------------------------------------------------------
__global__ void ggnn_pack(const float* __restrict__ src, __bf16* __restrict__ dst,
                          int ncols, int mode) {
  int tid = blockIdx.x * blockDim.x + threadIdx.x;
  int lane = tid & 31;
  int fid = tid >> 5;
  int nfrag = (ncols / 16) * GG_KT;
  if (fid >= nfrag) return;
  int kt = fid % GG_KT;
  int nt = fid / GG_KT;
  int khalf = (lane < 16) ? 0 : 8;
  int n = nt * 16 + (lane & 15);
  __bf16* out = dst + ((size_t)fid * 32 + lane) * 16;
#pragma unroll
  for (int j = 0; j < 16; ++j) {
    int k = kt * 32 + ((j < 8) ? (khalf + j) : (16 + khalf + (j - 8)));
    float v;
    if (mode == 0) {
      int e = n >> 8;        // n / 256
      int c = n & 255;       // n % 256
      v = src[((size_t)e * GG_H + k) * GG_H + c];
    } else {
      v = src[(size_t)n * GG_H + k];
    }
    out[j] = (__bf16)v;
  }
}

// ---------------------------------------------------------------------------
// h0 = emb[x]  (fp32 master copy + bf16 matmul copy)
// ---------------------------------------------------------------------------
__global__ void ggnn_embed(const int* __restrict__ x, const float* __restrict__ emb,
                           float* __restrict__ hf, __bf16* __restrict__ hb) {
  size_t tid = (size_t)blockIdx.x * blockDim.x + threadIdx.x;
  if (tid >= (size_t)GG_BS * GG_H) return;
  size_t r = tid >> 8;
  int c = (int)(tid & 255);
  float v = emb[(size_t)x[r] * GG_H + c];
  hf[tid] = v;
  hb[tid] = (__bf16)v;
}

// ---------------------------------------------------------------------------
// m = shift_fwd(h@We0+b0) + shift_bwd(h@We1+b1) + (h@We2+b2), fused into one
// accumulator per 16x16 tile. Sequence shift handled by offsetting A rows;
// boundary rows get zero A-fragments AND drop the corresponding bias.
// Block = 4 waves sharing ONE column tile -> edge-weight B fragments staged
// to LDS once per block via async global->LDS copies.
// ---------------------------------------------------------------------------
__global__ void __launch_bounds__(128) ggnn_message(
    const __bf16* __restrict__ hb, const __bf16* __restrict__ Pe,
    const float* __restrict__ b_edge, __bf16* __restrict__ mb) {
  __shared__ __align__(16) __bf16 sB[3 * GG_FRAG_H];  // 24 KB
  int lane = threadIdx.x & 31;
  int wave = threadIdx.x >> 5;
  int nt = blockIdx.x % GG_NT_H;                 // column tile shared by block
  int mt = (blockIdx.x / GG_NT_H) * 4 + wave;    // row tile per wave

  const __bf16* src[3] = {Pe + (size_t)(0 * GG_NT_H + nt) * GG_FRAG_H,
                          Pe + (size_t)(1 * GG_NT_H + nt) * GG_FRAG_H,
                          Pe + (size_t)(2 * GG_NT_H + nt) * GG_FRAG_H};
  stage_regions<3>(src, sB, threadIdx.x);
  cp_wait_all();

  int r0 = mt * 16;
  int s0 = r0 & (GG_S - 1);          // S divides 16-row tiles -> one sequence/tile
  int lrow = r0 + (lane & 15);
  int ls = s0 + (lane & 15);
  bool fwd_ok = (ls >= 1);
  bool bwd_ok = (ls < GG_S - 1);

  v8f acc = {0.f, 0.f, 0.f, 0.f, 0.f, 0.f, 0.f, 0.f};
#pragma unroll
  for (int kt = 0; kt < GG_KT; ++kt) {
    int k0 = kt * 32;
    v16bf a_fwd  = ld_a_frag(hb, lrow - 1, k0, lane, fwd_ok);
    v16bf a_bwd  = ld_a_frag(hb, lrow + 1, k0, lane, bwd_ok);
    v16bf a_self = ld_a_frag(hb, lrow,     k0, lane, true);
    acc = wmma_bf16(a_fwd,  ld_b_lds(sB, 0, kt, lane), acc);
    acc = wmma_bf16(a_bwd,  ld_b_lds(sB, 1, kt, lane), acc);
    acc = wmma_bf16(a_self, ld_b_lds(sB, 2, kt, lane), acc);
  }

  int col = nt * 16 + (lane & 15);
  float be0 = b_edge[col];
  float be1 = b_edge[GG_H + col];
  float be2 = b_edge[2 * GG_H + col];
  int rbase = r0 + ((lane < 16) ? 0 : 8);
  int sbase = s0 + ((lane < 16) ? 0 : 8);
#pragma unroll
  for (int i = 0; i < 8; ++i) {
    int rr = rbase + i;
    int ss = sbase + i;
    float bias = be2 + ((ss > 0) ? be0 : 0.f) + ((ss < GG_S - 1) ? be1 : 0.f);
    mb[(size_t)rr * GG_H + col] = (__bf16)(acc[i] + bias);
  }
}

// ---------------------------------------------------------------------------
// Fused GRU step: gi = m@W_ih^T, gh = h@W_hh^T (6 gate tiles, 48 WMMAs per
// 16x16 output tile) + sigmoid/tanh gate math; writes h_new fp32 + bf16.
// Block = 4 waves sharing ONE column tile -> 48 KB of gate weights staged to
// LDS once per block.
// ---------------------------------------------------------------------------
__global__ void __launch_bounds__(128) ggnn_gru(
    const __bf16* __restrict__ mb, const __bf16* __restrict__ hb,
    const float* __restrict__ hf, const __bf16* __restrict__ Pih,
    const __bf16* __restrict__ Phh, const float* __restrict__ b_ih,
    const float* __restrict__ b_hh, float* __restrict__ hf_out,
    __bf16* __restrict__ hb_out) {
  __shared__ __align__(16) __bf16 sB[6 * GG_FRAG_H];  // 48 KB
  int lane = threadIdx.x & 31;
  int wave = threadIdx.x >> 5;
  int ct = blockIdx.x % GG_NT_H;                 // column tile shared by block
  int mt = (blockIdx.x / GG_NT_H) * 4 + wave;    // row tile per wave

  const __bf16* src[6] = {Pih + (size_t)(0 * GG_NT_H + ct) * GG_FRAG_H,
                          Pih + (size_t)(1 * GG_NT_H + ct) * GG_FRAG_H,
                          Pih + (size_t)(2 * GG_NT_H + ct) * GG_FRAG_H,
                          Phh + (size_t)(0 * GG_NT_H + ct) * GG_FRAG_H,
                          Phh + (size_t)(1 * GG_NT_H + ct) * GG_FRAG_H,
                          Phh + (size_t)(2 * GG_NT_H + ct) * GG_FRAG_H};
  stage_regions<6>(src, sB, threadIdx.x);
  cp_wait_all();

  int r0 = mt * 16;
  int lrow = r0 + (lane & 15);

  v8f ir = {0.f, 0.f, 0.f, 0.f, 0.f, 0.f, 0.f, 0.f};
  v8f iz = ir, inn = ir, hr = ir, hz = ir, hn = ir;
#pragma unroll
  for (int kt = 0; kt < GG_KT; ++kt) {
    int k0 = kt * 32;
    v16bf am = ld_a_frag(mb, lrow, k0, lane, true);
    v16bf ah = ld_a_frag(hb, lrow, k0, lane, true);
    ir  = wmma_bf16(am, ld_b_lds(sB, 0, kt, lane), ir);
    iz  = wmma_bf16(am, ld_b_lds(sB, 1, kt, lane), iz);
    inn = wmma_bf16(am, ld_b_lds(sB, 2, kt, lane), inn);
    hr  = wmma_bf16(ah, ld_b_lds(sB, 3, kt, lane), hr);
    hz  = wmma_bf16(ah, ld_b_lds(sB, 4, kt, lane), hz);
    hn  = wmma_bf16(ah, ld_b_lds(sB, 5, kt, lane), hn);
  }

  int c = ct * 16 + (lane & 15);
  float bir = b_ih[c], biz = b_ih[GG_H + c], bin = b_ih[2 * GG_H + c];
  float bhr = b_hh[c], bhz = b_hh[GG_H + c], bhn = b_hh[2 * GG_H + c];
  int rbase = r0 + ((lane < 16) ? 0 : 8);
#pragma unroll
  for (int i = 0; i < 8; ++i) {
    int rr = rbase + i;
    float rg = 1.f / (1.f + __expf(-(ir[i] + bir + hr[i] + bhr)));
    float zg = 1.f / (1.f + __expf(-(iz[i] + biz + hz[i] + bhz)));
    float ng = tanhf(inn[i] + bin + rg * (hn[i] + bhn));
    float ho = hf[(size_t)rr * GG_H + c];
    float hv = (1.f - zg) * ng + zg * ho;
    hf_out[(size_t)rr * GG_H + c] = hv;
    hb_out[(size_t)rr * GG_H + c] = (__bf16)hv;
  }
}

// ---------------------------------------------------------------------------
// out[b, v] = h_final[b, S-1, :] . out_w[v, :] + out_b[v]   (64 x 32000)
// Block = 4 waves (the 4 row tiles of M=64) sharing one 8 KB B slice in LDS.
// ---------------------------------------------------------------------------
__global__ void __launch_bounds__(128) ggnn_out(
    const __bf16* __restrict__ hb, const __bf16* __restrict__ Pout,
    const float* __restrict__ out_b, float* __restrict__ out) {
  __shared__ __align__(16) __bf16 sB[GG_FRAG_H];  // 8 KB
  int lane = threadIdx.x & 31;
  int wave = threadIdx.x >> 5;
  int nt = blockIdx.x;   // 0..1999 column tiles, shared by block
  int mt = wave;         // 0..3 row tiles (M = 64)

  const __bf16* src[1] = {Pout + (size_t)nt * GG_FRAG_H};
  stage_regions<1>(src, sB, threadIdx.x);
  cp_wait_all();

  int bidx = mt * 16 + (lane & 15);
  int lrow = bidx * GG_S + (GG_S - 1);  // last position of each sequence

  v8f acc = {0.f, 0.f, 0.f, 0.f, 0.f, 0.f, 0.f, 0.f};
#pragma unroll
  for (int kt = 0; kt < GG_KT; ++kt) {
    v16bf a = ld_a_frag(hb, lrow, kt * 32, lane, true);
    acc = wmma_bf16(a, ld_b_lds(sB, 0, kt, lane), acc);
  }
  int v = nt * 16 + (lane & 15);
  float bv = out_b[v];
  int bbase = mt * 16 + ((lane < 16) ? 0 : 8);
#pragma unroll
  for (int i = 0; i < 8; ++i) {
    out[(size_t)(bbase + i) * GG_V + v] = acc[i] + bv;
  }
}

// ---------------------------------------------------------------------------
extern "C" void kernel_launch(void* const* d_in, const int* in_sizes, int n_in,
                              void* d_out, int out_size, void* d_ws, size_t ws_size,
                              hipStream_t stream) {
  const int*   x      = (const int*)d_in[0];
  const float* emb    = (const float*)d_in[1];
  const float* W_edge = (const float*)d_in[2];
  const float* b_edge = (const float*)d_in[3];
  const float* W_ih   = (const float*)d_in[4];
  const float* W_hh   = (const float*)d_in[5];
  const float* b_ih   = (const float*)d_in[6];
  const float* b_hh   = (const float*)d_in[7];
  const float* out_w  = (const float*)d_in[8];
  const float* out_b  = (const float*)d_in[9];
  float* out = (float*)d_out;

  // Workspace carve-up (~129 MiB total)
  char* p = (char*)d_ws;
  const size_t nh = (size_t)GG_BS * GG_H;
  float*  hfA  = (float*)p;  p += nh * sizeof(float);
  float*  hfB  = (float*)p;  p += nh * sizeof(float);
  __bf16* hbA  = (__bf16*)p; p += nh * sizeof(__bf16);
  __bf16* hbB  = (__bf16*)p; p += nh * sizeof(__bf16);
  __bf16* mb   = (__bf16*)p; p += nh * sizeof(__bf16);
  __bf16* Pe   = (__bf16*)p; p += (size_t)GG_H * 3 * GG_H * sizeof(__bf16);
  __bf16* Pih  = (__bf16*)p; p += (size_t)GG_H * 3 * GG_H * sizeof(__bf16);
  __bf16* Phh  = (__bf16*)p; p += (size_t)GG_H * 3 * GG_H * sizeof(__bf16);
  __bf16* Pout = (__bf16*)p; p += (size_t)GG_H * GG_V * sizeof(__bf16);

  // One-time weight repack + embedding gather (deterministic every call)
  ggnn_pack<<<48, 256, 0, stream>>>(W_edge, Pe, 3 * GG_H, 0);
  ggnn_pack<<<48, 256, 0, stream>>>(W_ih, Pih, 3 * GG_H, 1);
  ggnn_pack<<<48, 256, 0, stream>>>(W_hh, Phh, 3 * GG_H, 1);
  ggnn_pack<<<2000, 256, 0, stream>>>(out_w, Pout, GG_V, 1);
  ggnn_embed<<<(int)((nh + 255) / 256), 256, 0, stream>>>(x, emb, hfA, hbA);

  float*  hf_cur = hfA; float*  hf_nxt = hfB;
  __bf16* hb_cur = hbA; __bf16* hb_nxt = hbB;
  const int nblk = (GG_MT * GG_NT_H) / 4;  // 8192 blocks x 4 waves
  for (int step = 0; step < GG_STEPS; ++step) {
    ggnn_message<<<nblk, 128, 0, stream>>>(hb_cur, Pe, b_edge, mb);
    ggnn_gru<<<nblk, 128, 0, stream>>>(mb, hb_cur, hf_cur, Pih, Phh, b_ih, b_hh,
                                       hf_nxt, hb_nxt);
    float*  tf = hf_cur; hf_cur = hf_nxt; hf_nxt = tf;
    __bf16* tb = hb_cur; hb_cur = hb_nxt; hb_nxt = tb;
  }

  ggnn_out<<<GG_V / 16, 128, 0, stream>>>(hb_cur, Pout, out_b, out);
}
